// SpatialTransformer_21672404975844
// MI455X (gfx1250) — matrix operand order repaired and checked by reading, not compiled, not run on hardware
//
#include <hip/hip_runtime.h>
#include <hip/hip_bf16.h>
#include <math.h>

typedef __attribute__((ext_vector_type(16))) __bf16           v16bf;
typedef __attribute__((ext_vector_type(16))) unsigned short   v16us;
typedef __attribute__((ext_vector_type(8)))  float            v8f;

__device__ __forceinline__ unsigned short f2bf(float f) {
    union { float f; unsigned int u; } v; v.f = f;
    unsigned int u = v.u;
    unsigned int r = (u + 0x7FFFu + ((u >> 16) & 1u)) >> 16;  // RNE
    return (unsigned short)r;
}

// ---------------------------------------------------------------------------
// Generic bf16-WMMA GEMM:  C[M,N] (f32) = act( A*B + bias ) [+ residual][+=C]
//   A element (m,k) = transA ? A[k*lda+m] : A[m*lda+k]
//   B element (k,n) = transB ? B[n*ldb+k] : B[k*ldb+n]
// Tile 64x64, K-step 32, 4 waves (128 threads), 2x2 16x16 frags per wave.
// Non-transposed operands are staged global->LDS with CDNA5 async-LDS copies
// (ASYNCcnt) into double buffers; transposed operands use sync bf16 staging.
// ---------------------------------------------------------------------------
#define TM 64
#define TN 64
#define TK 32

__global__ __launch_bounds__(128)
void gemm_wmma_bf16(const float* __restrict__ A, long long sA, int lda, int transA,
                    const float* __restrict__ B, long long sB, int ldb, int transB,
                    float* __restrict__ C, long long sC, int ldc,
                    const float* __restrict__ bias,
                    const float* __restrict__ res, long long sR, int ldr,
                    int M, int N, int K, int accumulate, int act)
{
    // f32 staging buffers (async path); bf16 sync path aliases the same memory
    __shared__ float AsF[2][TM][TK];     // 2 x 8KB
    __shared__ float BsF[2][TK][TN];     // 2 x 8KB

    int batch = blockIdx.z;
    A += (long long)batch * sA;
    B += (long long)batch * sB;
    C += (long long)batch * sC;
    if (res) res += (long long)batch * sR;

    const int m0   = blockIdx.y * TM;
    const int n0   = blockIdx.x * TN;
    const int tid  = threadIdx.x;
    const int lane = tid & 31;
    const int wave = tid >> 5;
    const int wm   = (wave >> 1) * 32;
    const int wn   = (wave & 1) * 32;
    const int col  = lane & 15;
    const int hi   = lane >> 4;

    const bool fullA = ((M & (TM - 1)) == 0) && ((K & (TK - 1)) == 0);
    const bool fullB = ((N & (TN - 1)) == 0) && ((K & (TK - 1)) == 0);

    // stage one K-step tile pair into buffer `buf`
    auto stage = [&](int buf, int kb) {
        // ---- A tile ----
        if (transA) {                       // sync bf16, transpose-on-load
            unsigned short* Ah = (unsigned short*)&AsF[buf][0][0];
            for (int idx = tid; idx < TM * TK; idx += 128) {
                int m = idx >> 5, k = idx & (TK - 1);
                int gm = m0 + m, gk = kb + k;
                float v = 0.0f;
                if (gm < M && gk < K) v = A[(long long)gk * lda + gm];
                Ah[m * TK + k] = f2bf(v);
            }
        } else if (fullA) {                 // async f32, 16B per lane-chunk
            #pragma unroll
            for (int c = 0; c < 4; c++) {
                int idx = tid + c * 128;            // 512 chunks of 16B
                int m  = idx >> 3;
                int c4 = (idx & 7) << 2;
                unsigned lds = (unsigned)(size_t)&AsF[buf][m][c4];
                const float* gp = A + (long long)(m0 + m) * lda + kb + c4;
                asm volatile("global_load_async_to_lds_b128 %0, %1, off"
                             :: "v"(lds), "v"((unsigned long long)(size_t)gp)
                             : "memory");
            }
        } else {                            // sync f32 fallback (guarded)
            for (int idx = tid; idx < TM * TK; idx += 128) {
                int m = idx >> 5, k = idx & (TK - 1);
                int gm = m0 + m, gk = kb + k;
                AsF[buf][m][k] = (gm < M && gk < K) ? A[(long long)gm * lda + gk] : 0.0f;
            }
        }
        // ---- B tile ----
        if (transB) {
            unsigned short* Bh = (unsigned short*)&BsF[buf][0][0];
            for (int idx = tid; idx < TK * TN; idx += 128) {
                int k = idx >> 6, n = idx & (TN - 1);
                int gk = kb + k, gn = n0 + n;
                float v = 0.0f;
                if (gk < K && gn < N) v = B[(long long)gn * ldb + gk];
                Bh[k * TN + n] = f2bf(v);
            }
        } else if (fullB) {
            #pragma unroll
            for (int c = 0; c < 4; c++) {
                int idx = tid + c * 128;            // 512 chunks of 16B
                int k  = idx >> 4;
                int c4 = (idx & 15) << 2;
                unsigned lds = (unsigned)(size_t)&BsF[buf][k][c4];
                const float* gp = B + (long long)(kb + k) * ldb + n0 + c4;
                asm volatile("global_load_async_to_lds_b128 %0, %1, off"
                             :: "v"(lds), "v"((unsigned long long)(size_t)gp)
                             : "memory");
            }
        } else {
            for (int idx = tid; idx < TK * TN; idx += 128) {
                int k = idx >> 6, n = idx & (TN - 1);
                int gk = kb + k, gn = n0 + n;
                BsF[buf][k][n] = (gk < K && gn < N) ? B[(long long)gk * ldb + gn] : 0.0f;
            }
        }
    };

    v8f acc[2][2];
    acc[0][0] = (v8f){}; acc[0][1] = (v8f){};
    acc[1][0] = (v8f){}; acc[1][1] = (v8f){};

    const int nk = (K + TK - 1) / TK;

    stage(0, 0);
    asm volatile("s_wait_asynccnt 0x0" ::: "memory");
    __syncthreads();

    for (int ib = 0; ib < nk; ib++) {
        int buf = ib & 1;
        // prefetch-ahead: fill the next buffer while this one computes
        if (ib + 1 < nk) stage(buf ^ 1, (ib + 1) * TK);
        // L2 warm-up two steps ahead for the (transposed) weight operand
        if (transB && (ib + 2) < nk) {
            int gn = n0 + (tid & 63);
            if (gn < N) __builtin_prefetch(&B[(long long)gn * ldb + (ib + 2) * TK], 0, 1);
        }

        const unsigned short* Ah = (const unsigned short*)&AsF[buf][0][0];
        const unsigned short* Bh = (const unsigned short*)&BsF[buf][0][0];

        v16bf afrag[2], bfrag[2];
        #pragma unroll
        for (int i = 0; i < 2; i++) {
            int row = wm + i * 16 + col;
            v16us a;
            #pragma unroll
            for (int e = 0; e < 16; e++) {
                int kk = ((e >> 3) << 4) + (hi << 3) + (e & 7);   // 16-bit A layout
                a[e] = transA ? Ah[row * TK + kk] : f2bf(AsF[buf][row][kk]);
            }
            afrag[i] = __builtin_bit_cast(v16bf, a);
        }
        #pragma unroll
        for (int j = 0; j < 2; j++) {
            int cn = wn + j * 16 + col;
            v16us b;
            #pragma unroll
            for (int e = 0; e < 16; e++) {
                int kk = ((e >> 3) << 4) + (hi << 3) + (e & 7);   // 16-bit B layout
                b[e] = transB ? Bh[kk * TN + cn] : f2bf(BsF[buf][kk][cn]);
            }
            bfrag[j] = __builtin_bit_cast(v16bf, b);
        }
        #pragma unroll
        for (int i = 0; i < 2; i++)
            #pragma unroll
            for (int j = 0; j < 2; j++)
                acc[i][j] = __builtin_amdgcn_wmma_f32_16x16x32_bf16(
                    false, afrag[i], false, bfrag[j], (short)0, acc[i][j], false, false);

        // make next buffer's async fills visible, release this buffer
        asm volatile("s_wait_asynccnt 0x0" ::: "memory");
        __syncthreads();
    }

    // epilogue: accumulate / bias / activation / residual
    #pragma unroll
    for (int i = 0; i < 2; i++) {
        #pragma unroll
        for (int j = 0; j < 2; j++) {
            int gn = n0 + wn + j * 16 + col;
            #pragma unroll
            for (int r = 0; r < 8; r++) {
                int gm = m0 + wm + i * 16 + r + (hi << 3);
                if (gm < M && gn < N) {
                    float v = acc[i][j][r];
                    if (accumulate) v += C[(long long)gm * ldc + gn];
                    if (bias)       v += bias[gn];
                    if (act == 1)   v = 0.5f * v * (1.0f + erff(v * 0.70710678118654752f)); // exact GELU
                    if (res)        v += res[(long long)gm * ldr + gn];
                    C[(long long)gm * ldc + gn] = v;
                }
            }
        }
    }
}

// ---------------------------------------------------------------------------
// Fused multi-head attention. qkv: [48*512, 192] (q|k|v, head-major inside 64)
// One workgroup = (bt,head) x 32 query rows; 2 waves, 16 q-rows per wave.
// out: [48*512, 64]  (heads concatenated)
// ---------------------------------------------------------------------------
#define NCTX 512

__global__ __launch_bounds__(64)
void attn_wmma(const float* __restrict__ qkv, float* __restrict__ out)
{
    __shared__ unsigned short Ks[NCTX][8];
    __shared__ unsigned short Vs[NCTX][8];
    __shared__ float S[2][16][NCTX + 8];
    __shared__ float Inv[2][16];

    int bh   = blockIdx.y;
    int bt   = bh >> 3;
    int head = bh & 7;
    int tid  = threadIdx.x;
    int lane = tid & 31;
    int wave = tid >> 5;
    int col  = lane & 15;
    int hi   = lane >> 4;
    const float scale = 0.35355339059327373f;   // 8^-0.5

    const float* base = qkv + (long long)bt * NCTX * 192 + head * 8;

    // stage K, V (512 x 8) as bf16
    for (int idx = tid; idx < NCTX * 8; idx += 64) {
        int j = idx >> 3, d = idx & 7;
        Ks[j][d] = f2bf(base[(long long)j * 192 + 64  + d]);
        Vs[j][d] = f2bf(base[(long long)j * 192 + 128 + d]);
    }
    __syncthreads();

    int r0 = blockIdx.x * 32 + wave * 16;       // first query row of this wave

    // Q fragment: 16 rows x K(=32, hd=8 zero-padded)
    v16us aq;
    #pragma unroll
    for (int e = 0; e < 16; e++) {
        int kk = ((e >> 3) << 4) + (hi << 3) + (e & 7);
        aq[e] = (kk < 8) ? f2bf(base[(long long)(r0 + col) * 192 + kk]) : (unsigned short)0;
    }
    v16bf qf = __builtin_bit_cast(v16bf, aq);

    // scores S = scale * Q K^T  (32 column-blocks of 16)
    for (int nb = 0; nb < NCTX / 16; nb++) {
        v16us bu;
        #pragma unroll
        for (int e = 0; e < 16; e++) {
            int kk = ((e >> 3) << 4) + (hi << 3) + (e & 7);
            bu[e] = (kk < 8) ? Ks[nb * 16 + col][kk] : (unsigned short)0;
        }
        v16bf bf = __builtin_bit_cast(v16bf, bu);
        v8f sc = (v8f){};
        sc = __builtin_amdgcn_wmma_f32_16x16x32_bf16(false, qf, false, bf, (short)0, sc, false, false);
        #pragma unroll
        for (int r = 0; r < 8; r++)
            S[wave][r + (hi << 3)][nb * 16 + col] = sc[r] * scale;
    }
    __syncthreads();

    // softmax: lane handles row=col, half hi (256 cols)
    {
        int row = col;
        float mx = -1e30f;
        for (int c2 = 0; c2 < 256; c2++)
            mx = fmaxf(mx, S[wave][row][(hi << 8) + c2]);
        mx = fmaxf(mx, __shfl_xor(mx, 16, 32));
        float sum = 0.0f;
        for (int c2 = 0; c2 < 256; c2++) {
            float e = __expf(S[wave][row][(hi << 8) + c2] - mx);
            S[wave][row][(hi << 8) + c2] = e;   // unnormalized P
            sum += e;
        }
        sum += __shfl_xor(sum, 16, 32);
        if (hi == 0) Inv[wave][row] = 1.0f / sum;
    }
    __syncthreads();

    // O = P V  (K = 512 in 16 steps of 32), fold 1/sum on output
    v8f o = (v8f){};
    for (int jb = 0; jb < NCTX / 32; jb++) {
        v16us au, bu;
        #pragma unroll
        for (int e = 0; e < 16; e++) {
            int kk = ((e >> 3) << 4) + (hi << 3) + (e & 7);
            au[e] = f2bf(S[wave][col][jb * 32 + kk]);
        }
        #pragma unroll
        for (int e = 0; e < 16; e++) {
            int kk = ((e >> 3) << 4) + (hi << 3) + (e & 7);
            bu[e] = (col < 8) ? Vs[jb * 32 + kk][col] : (unsigned short)0;
        }
        v16bf af = __builtin_bit_cast(v16bf, au);
        v16bf bf = __builtin_bit_cast(v16bf, bu);
        o = __builtin_amdgcn_wmma_f32_16x16x32_bf16(false, af, false, bf, (short)0, o, false, false);
    }

    if (col < 8) {
        #pragma unroll
        for (int r = 0; r < 8; r++) {
            int rowg = r0 + r + (hi << 3);
            out[((long long)(bt * NCTX + rowg)) * 64 + head * 8 + col]
                = o[r] * Inv[wave][r + (hi << 3)];
        }
    }
}

// ---------------------------------------------------------------------------
// LayerNorm over c=64, one wave per row (wave32: 2 elements/lane)
// ---------------------------------------------------------------------------
__global__ __launch_bounds__(256)
void ln_kernel(const float* __restrict__ h, const float* __restrict__ g,
               const float* __restrict__ bta, float* __restrict__ y, int rows)
{
    int wave = threadIdx.x >> 5, lane = threadIdx.x & 31;
    int row = blockIdx.x * 8 + wave;
    if (row >= rows) return;
    const float* x = h + (long long)row * 64;
    float x0 = x[lane], x1 = x[lane + 32];
    float s = x0 + x1;
    for (int o = 16; o; o >>= 1) s += __shfl_xor(s, o, 32);
    float mean = s * (1.0f / 64.0f);
    float d0 = x0 - mean, d1 = x1 - mean;
    float v = d0 * d0 + d1 * d1;
    for (int o = 16; o; o >>= 1) v += __shfl_xor(v, o, 32);
    float rstd = rsqrtf(v * (1.0f / 64.0f) + 1e-5f);
    y[(long long)row * 64 + lane]      = d0 * rstd * g[lane]      + bta[lane];
    y[(long long)row * 64 + lane + 32] = d1 * rstd * g[lane + 32] + bta[lane + 32];
}

// ---------------------------------------------------------------------------
// Layout permutes.  x:[4,64,512,12]  h:[48,512,64]  (bt = b*12+t)
// ---------------------------------------------------------------------------
__global__ void permute_in(const float* __restrict__ x, const float* __restrict__ pos,
                           float* __restrict__ h, int total)
{
    int idx = blockIdx.x * blockDim.x + threadIdx.x;
    if (idx >= total) return;
    int c  = idx & 63;
    int n  = (idx >> 6) & 511;
    int bt = idx >> 15;
    int b  = bt / 12, t = bt - b * 12;
    h[idx] = x[(((long long)b * 64 + c) * 512 + n) * 12 + t] + pos[n * 64 + c];
}

__global__ void permute_out(const float* __restrict__ h, float* __restrict__ out, int total)
{
    int idx = blockIdx.x * blockDim.x + threadIdx.x;
    if (idx >= total) return;
    int t = idx % 12;
    int n = (idx / 12) % 512;
    int c = (idx / (12 * 512)) % 64;
    int b = idx / (12 * 512 * 64);
    out[idx] = h[(((long long)(b * 12 + t)) * 512 + n) * 64 + c];
}

// ---------------------------------------------------------------------------
static inline void gemm(hipStream_t s,
                        const float* A, long long sA, int lda, int tA,
                        const float* B, long long sB, int ldb, int tB,
                        float* C, long long sC, int ldc,
                        const float* bias, const float* res, long long sR, int ldr,
                        int M, int N, int K, int batch, int accum, int act)
{
    dim3 grid((N + TN - 1) / TN, (M + TM - 1) / TM, batch);
    gemm_wmma_bf16<<<grid, 128, 0, s>>>(A, sA, lda, tA, B, sB, ldb, tB,
                                        C, sC, ldc, bias, res, sR, ldr,
                                        M, N, K, accum, act);
}

extern "C" void kernel_launch(void* const* d_in, const int* in_sizes, int n_in,
                              void* d_out, int out_size, void* d_ws, size_t ws_size,
                              hipStream_t stream)
{
    const float* x      = (const float*)d_in[0];
    const float* adj    = (const float*)d_in[1];
    const float* pos    = (const float*)d_in[2];
    const float* qkv_w  = (const float*)d_in[3];
    const float* proj_w = (const float*)d_in[4];
    const float* proj_b = (const float*)d_in[5];
    const float* ln_g   = (const float*)d_in[6];
    const float* ln_b   = (const float*)d_in[7];
    const float* ff_w1  = (const float*)d_in[8];
    const float* ff_b1  = (const float*)d_in[9];
    const float* ff_w2  = (const float*)d_in[10];
    const float* ff_b2  = (const float*)d_in[11];
    const float* gcn_w  = (const float*)d_in[12];
    const float* gcn_b  = (const float*)d_in[13];
    float* out = (float*)d_out;

    const long long R = 48LL * 512;            // 24576 rows
    float* ws    = (float*)d_ws;
    float* h     = ws;                          // R*64
    float* resid = h     + R * 64;              // R*64
    float* qkvb  = resid + R * 64;              // R*192
    float* attb  = qkvb  + R * 192;             // R*64
    float* yb    = attb  + R * 64;              // R*64
    float* ffb   = yb    + R * 64;              // R*256
    float* x1b   = ffb   + R * 256;             // R*64
    float* x2b   = x1b   + R * 64;              // R*64

    const int total = (int)(R * 64);            // 1572864
    permute_in<<<dim3((total + 255) / 256), 256, 0, stream>>>(x, pos, h, total);

    for (int l = 0; l < 2; l++) {
        const float* qw  = qkv_w  + (long long)l * 192 * 64;
        const float* pw  = proj_w + (long long)l * 64 * 64;
        const float* pb  = proj_b + (long long)l * 64;
        const float* lg  = ln_g   + (long long)l * 64;
        const float* lb  = ln_b   + (long long)l * 64;
        const float* f1  = ff_w1  + (long long)l * 256 * 64;
        const float* fb1 = ff_b1  + (long long)l * 256;
        const float* f2  = ff_w2  + (long long)l * 64 * 256;
        const float* fb2 = ff_b2  + (long long)l * 64;
        const float* gw  = gcn_w  + (long long)l * 64 * 192;
        const float* gb  = gcn_b  + (long long)l * 64;

        // residual (layer input) for the GCN branch
        hipMemcpyAsync(resid, h, (size_t)R * 64 * sizeof(float),
                       hipMemcpyDeviceToDevice, stream);

        // qkv = h @ qkv_w^T          [R,192]
        gemm(stream, h, 0, 64, 0, qw, 0, 64, 1, qkvb, 0, 192,
             nullptr, nullptr, 0, 0, (int)R, 192, 64, 1, 0, 0);

        // fused MHA -> attb [R,64]
        attn_wmma<<<dim3(NCTX / 32, 48 * 8), 64, 0, stream>>>(qkvb, attb);

        // h = attb @ proj_w^T + proj_b + h
        gemm(stream, attb, 0, 64, 0, pw, 0, 64, 1, h, 0, 64,
             pb, h, 0, 64, (int)R, 64, 64, 1, 0, 0);

        // y = LN(h)
        ln_kernel<<<dim3((int)(R / 8)), 256, 0, stream>>>(h, lg, lb, yb, (int)R);

        // ffb = gelu(y @ ff_w1^T + b1)
        gemm(stream, yb, 0, 64, 0, f1, 0, 64, 1, ffb, 0, 256,
             fb1, nullptr, 0, 0, (int)R, 256, 64, 1, 0, 1);

        // h = ffb @ ff_w2^T + b2 + h
        gemm(stream, ffb, 0, 256, 0, f2, 0, 256, 1, h, 0, 64,
             fb2, h, 0, 64, (int)R, 64, 256, 1, 0, 0);

        // GCN diffusion: x1[bt] = adj^T @ resid[bt] ; x2[bt] = adj^T @ x1[bt]
        gemm(stream, adj, 0, 512, 1, resid, 512 * 64, 64, 0, x1b, 512 * 64, 64,
             nullptr, nullptr, 0, 0, 512, 64, 512, 48, 0, 0);
        gemm(stream, adj, 0, 512, 1, x1b, 512 * 64, 64, 0, x2b, 512 * 64, 64,
             nullptr, nullptr, 0, 0, 512, 64, 512, 48, 0, 0);

        // GCN 1x1 conv over concat [r|x1|x2]: three accumulating 64x64 GEMMs
        gemm(stream, resid, 0, 64, 0, gw, 0, 192, 1, h, 0, 64,
             gb, h, 0, 64, (int)R, 64, 64, 1, 0, 0);            // h = r W0^T + b + h
        gemm(stream, x1b, 0, 64, 0, gw + 64, 0, 192, 1, h, 0, 64,
             nullptr, nullptr, 0, 0, (int)R, 64, 64, 1, 1, 0);  // h += x1 W1^T
        gemm(stream, x2b, 0, 64, 0, gw + 128, 0, 192, 1, h, 0, 64,
             nullptr, nullptr, 0, 0, (int)R, 64, 64, 1, 1, 0);  // h += x2 W2^T
    }

    permute_out<<<dim3((total + 255) / 256), 256, 0, stream>>>(h, out, total);
}
